// Attention_22608707846241
// MI455X (gfx1250) — matrix operand order repaired and checked by reading, not compiled, not run on hardware
//
#include <hip/hip_runtime.h>
#include <hip/hip_bf16.h>

typedef __bf16 bf16;
typedef __attribute__((ext_vector_type(16))) __bf16 bf16x16;
typedef __attribute__((ext_vector_type(8)))  float  f32x8;

#define BATCH 64
#define SEQL  1024
#define KDIM  1024   // ENC == DEC
#define NDIM  1024   // ATT

#define MTILE 128
// Fragment-major LDS layout for the A tile:
//   addr = kblock*KBSTRIDE + s*SSTRIDE + c*CSTRIDE + perm(lane)*16  (+ j*2 within chunk)
// CSTRIDE = 528 B  (132 dw ≡ 4 mod 64  -> chunks skew banks)
// SSTRIDE = 1056 B (2 chunks)
// KBSTRIDE= 8512 B (2128 dw ≡ 16 mod 64 -> kblocks skew banks; 8 subtiles + 64B pad)
#define CSTRIDE  528
#define SSTRIDE  1056
#define KBSTRIDE 8512
#define SACC_OFF (32 * KBSTRIDE)    // 272384 B; +512 B of score accumulators

// ---------------------------------------------------------------------------
// Kernel P: W1 fp32 -> bf16 (one-time, into workspace)
// ---------------------------------------------------------------------------
__global__ void cvt_w1_bf16(const float* __restrict__ w1, bf16* __restrict__ w1b) {
  int idx = blockIdx.x * blockDim.x + threadIdx.x;      // one float4 each
  float4 v = ((const float4*)w1)[idx];
  union { bf16 h[4]; uint2 u; } p;
  p.h[0] = (bf16)v.x; p.h[1] = (bf16)v.y; p.h[2] = (bf16)v.z; p.h[3] = (bf16)v.w;
  ((uint2*)w1b)[idx] = p.u;
}

// ---------------------------------------------------------------------------
// Kernel A: proj_h[b,a] = hidden[b,:] @ W2[:,a] + W2_b[a]   (64 x 1024, tiny)
// ---------------------------------------------------------------------------
__global__ void proj_h_kernel(const float* __restrict__ hs, const float* __restrict__ w2,
                              const float* __restrict__ w2b, float* __restrict__ ph) {
  const int b = blockIdx.x, t = threadIdx.x;
  const float* hrow = hs + b * KDIM;
  float a0 = 0.f, a1 = 0.f, a2 = 0.f, a3 = 0.f;
  for (int e = 0; e < KDIM; ++e) {
    const float h = hrow[e];                 // wave-uniform -> scalar load
    const float* wr = w2 + (size_t)e * NDIM;
    a0 += h * wr[t];       a1 += h * wr[t + 256];
    a2 += h * wr[t + 512]; a3 += h * wr[t + 768];
  }
  float* o = ph + b * NDIM;
  o[t]       = a0 + w2b[t];       o[t + 256] = a1 + w2b[t + 256];
  o[t + 512] = a2 + w2b[t + 512]; o[t + 768] = a3 + w2b[t + 768];
}

// ---------------------------------------------------------------------------
// Kernel B: fused  scores[m] = V_w . tanh(features[m,:]@W1 + W1_b + proj_h[b,:]) + V_b
// 512 WGs x 256 threads; 128-row M-tile in LDS (bf16, fragment-major layout);
// 16 WMMA accumulators per wave.
// ---------------------------------------------------------------------------
__launch_bounds__(256)
__global__ void scores_kernel(const float* __restrict__ feat,
                              const bf16*  __restrict__ w1b16,
                              const float* __restrict__ w1bias,
                              const float* __restrict__ ph,
                              const float* __restrict__ vw,
                              const float* __restrict__ vb,
                              float* __restrict__ scores) {
  extern __shared__ char smem[];
  char*  tile = smem;
  float* sacc = (float*)(smem + SACC_OFF);

  const int tid = threadIdx.x;
  const int m0  = blockIdx.x * MTILE;     // 128 | 1024 -> tile is within one batch
  const int b   = m0 >> 10;

  if (tid < MTILE) sacc[tid] = 0.0f;

  // ---- stage features tile: fp32 global -> bf16 LDS, fragment-major.
  // A fragment (16x32 bf16): lanes 0-15 (h=0) hold K {0-7 | 16-23},
  // lanes 16-31 (h=1) hold K {8-15 | 24-31}; chunk c selects the 16B half.
  // Lane permutation perm(r,h) = 2r+h folds h into low bank bits.
  for (int idx = tid; idx < MTILE * 256; idx += 256) {
    const int row = idx >> 8, c4 = idx & 255;           // coalesced: wave = one row
    float4 v = ((const float4*)(feat + (size_t)(m0 + row) * KDIM))[c4];
    union { bf16 h[4]; uint2 u; } p;
    p.h[0] = (bf16)v.x; p.h[1] = (bf16)v.y; p.h[2] = (bf16)v.z; p.h[3] = (bf16)v.w;

    const int k0  = c4 * 4;              // first of 4 consecutive K
    const int kb  = k0 >> 5;             // kblock
    const int kin = k0 & 31;
    const int c   = (kin >> 4) & 1;      // 16B chunk select
    const int hb  = (kin >> 3) & 1;      // half-wave the data belongs to
    const int j   = kin & 7;             // 0 or 4: position within 8-elem chunk
    const int rr  = row & 15, ss = row >> 4;
    const int pl  = (rr << 1) | hb;      // perm(r + 16h)
    *(uint2*)(tile + kb * KBSTRIDE + ss * SSTRIDE + c * CSTRIDE + pl * 16 + j * 2) = p.u;
  }
  __syncthreads();

  const int wave = tid >> 5;
  const int lane = tid & 31;
  const int r    = lane & 15;      // N index within C fragment
  const int h    = lane >> 4;      // half-wave select
  const int pl   = ((lane & 15) << 1) | (lane >> 4);   // perm(lane)

  // Single per-lane A base; ALL 16 ds_loads of a K-step are immediate offsets
  // (max 3*KBSTRIDE + 7*SSTRIDE + CSTRIDE = 33456 < 65536), so the only
  // loop-carried address VALU is one pointer bump per unrolled body — no VALU
  // writes WMMA source registers, no hazard NOPs.
  const char* const abase = tile + pl * 16;

  for (int nc = 0; nc < 4; ++nc) {
    const int n0 = nc * 256 + wave * 32;   // this wave's 32 N-columns this chunk

    // Per-lane B base, unused: the K loop loads B through bcol instead,
    // one bf16 per element at a k-row stride of 2048 B.
    const char* const bp = (const char*)(w1b16 + (size_t)lane * NDIM + n0);

    f32x8 acc[8][2];
    #pragma unroll
    for (int s = 0; s < 8; ++s) {
      #pragma unroll
      for (int v = 0; v < 8; ++v) { acc[s][0][v] = 0.f; acc[s][1][v] = 0.f; }
    }

    #pragma unroll 4
    for (int kk = 0; kk < KDIM; kk += 32) {
      const char* const ak = abase + (kk >> 5) * KBSTRIDE;

      union { uint4 q[2]; bf16x16 v; } af[8];
      #pragma unroll
      for (int s = 0; s < 8; ++s) {
        af[s].q[0] = *(const uint4*)(ak + s * SSTRIDE);
        af[s].q[1] = *(const uint4*)(ak + s * SSTRIDE + CSTRIDE);
      }

      // B fragments (32x16 bf16): lane = N column n0+(lane&15), 16 K per lane in table order
      union { uint4 q[2]; bf16x16 v; } bf0, bf1;
      const bf16* bcol = w1b16 + (size_t)kk * NDIM + n0 + (lane & 15);
      #pragma unroll
      for (int i = 0; i < 16; ++i) {
        const int ki = (i < 8) ? (8 * h + i) : (16 + 8 * h + (i - 8));
        bf0.v[i] = bcol[(size_t)ki * NDIM];
        bf1.v[i] = bcol[(size_t)ki * NDIM + 16];
      }

      #pragma unroll
      for (int s = 0; s < 8; ++s) {
        acc[s][0] = __builtin_amdgcn_wmma_f32_16x16x32_bf16(
            false, af[s].v, false, bf0.v, (short)0, acc[s][0], false, false);
        acc[s][1] = __builtin_amdgcn_wmma_f32_16x16x32_bf16(
            false, af[s].v, false, bf1.v, (short)0, acc[s][1], false, false);
      }
    }

    // ---- epilogue: +bias +proj_h, tanh, *V_w, reduce over this wave's 16 N per tile
    #pragma unroll
    for (int j = 0; j < 2; ++j) {
      const int   n    = n0 + j * 16 + r;          // C layout: N = lane&15
      const float addv = w1bias[n] + ph[b * NDIM + n];
      const float vwn  = vw[n];
      #pragma unroll
      for (int s = 0; s < 8; ++s) {
        #pragma unroll
        for (int v = 0; v < 8; ++v) {
          const float x = acc[s][j][v] + addv;
          const float e = __expf(2.0f * x);        // NaN-safe tanh: 1 - 2/(e^{2x}+1)
          float val = (1.0f - 2.0f / (e + 1.0f)) * vwn;
          val += __shfl_xor(val, 1);               // reduce across 16-lane half
          val += __shfl_xor(val, 2);
          val += __shfl_xor(val, 4);
          val += __shfl_xor(val, 8);
          if (r == 0) atomicAdd(&sacc[s * 16 + v + 8 * h], val);  // C: M = v + 8h
        }
      }
    }
  }
  __syncthreads();
  if (tid < MTILE) scores[m0 + tid] = sacc[tid] + vb[0];
}

// ---------------------------------------------------------------------------
// Kernel C: softmax over L per batch row
// ---------------------------------------------------------------------------
__global__ void softmax_kernel(const float* __restrict__ scores, float* __restrict__ alpha) {
  __shared__ float red[256];
  const int b = blockIdx.x, t = threadIdx.x;
  const float* s = scores + b * SEQL;
  float v0 = s[t], v1 = s[t + 256], v2 = s[t + 512], v3 = s[t + 768];
  float m = fmaxf(fmaxf(v0, v1), fmaxf(v2, v3));
  red[t] = m; __syncthreads();
  for (int o = 128; o > 0; o >>= 1) { if (t < o) red[t] = fmaxf(red[t], red[t + o]); __syncthreads(); }
  m = red[0]; __syncthreads();
  v0 = __expf(v0 - m); v1 = __expf(v1 - m); v2 = __expf(v2 - m); v3 = __expf(v3 - m);
  red[t] = v0 + v1 + v2 + v3; __syncthreads();
  for (int o = 128; o > 0; o >>= 1) { if (t < o) red[t] += red[t + o]; __syncthreads(); }
  const float inv = 1.0f / red[0];
  float* a = alpha + b * SEQL;
  a[t] = v0 * inv; a[t + 256] = v1 * inv; a[t + 512] = v2 * inv; a[t + 768] = v3 * inv;
}

// ---------------------------------------------------------------------------
// Kernel D: context[b,e] = sum_l alpha[b,l] * features[b,l,e]  (bandwidth bound)
// ---------------------------------------------------------------------------
__global__ void context_kernel(const float* __restrict__ alpha, const float* __restrict__ feat,
                               float* __restrict__ ctx) {
  __shared__ float al[SEQL];
  const int b = blockIdx.x >> 2, ec = blockIdx.x & 3, t = threadIdx.x;
  for (int j = t; j < SEQL; j += 256) al[j] = alpha[b * SEQL + j];
  __syncthreads();
  const int e = ec * 256 + t;
  const float* f = feat + (size_t)b * SEQL * KDIM + e;
  float acc = 0.f;
  #pragma unroll 4
  for (int l = 0; l < SEQL; ++l) acc += al[l] * f[(size_t)l * KDIM];
  ctx[b * KDIM + e] = acc;
}

// ---------------------------------------------------------------------------
extern "C" void kernel_launch(void* const* d_in, const int* in_sizes, int n_in,
                              void* d_out, int out_size, void* d_ws, size_t ws_size,
                              hipStream_t stream) {
  const float* feat = (const float*)d_in[0];   // [64,1024,1024]
  const float* hs   = (const float*)d_in[1];   // [64,1024]
  const float* w1w  = (const float*)d_in[2];   // [1024,1024]
  const float* w1b  = (const float*)d_in[3];   // [1024]
  const float* w2w  = (const float*)d_in[4];   // [1024,1024]
  const float* w2b  = (const float*)d_in[5];   // [1024]
  const float* vww  = (const float*)d_in[6];   // [1024]
  const float* vb   = (const float*)d_in[7];   // [1]

  char*  ws     = (char*)d_ws;
  float* ph     = (float*)ws;                  // 256 KB
  float* scores = (float*)(ws + 262144);       // 256 KB
  bf16*  w1b16  = (bf16*)(ws + 524288);        // 2 MB

  float* alpha = (float*)d_out;                // [64,1024]
  float* ctx   = (float*)d_out + BATCH * SEQL; // [64,1024]

  const size_t lds = (size_t)SACC_OFF + MTILE * sizeof(float);  // ~266.5 KB
  hipFuncSetAttribute(reinterpret_cast<const void*>(scores_kernel),
                      hipFuncAttributeMaxDynamicSharedMemorySize, (int)lds);

  cvt_w1_bf16   <<<1024, 256, 0,  stream>>>(w1w, w1b16);
  proj_h_kernel <<<64,   256, 0,  stream>>>(hs, w2w, w2b, ph);
  scores_kernel <<<512,  256, lds, stream>>>(feat, w1b16, w1b, ph, vww, vb, scores);
  softmax_kernel<<<64,   256, 0,  stream>>>(scores, alpha);
  context_kernel<<<256,  256, 0,  stream>>>(alpha, feat, ctx);
}
